// TransformerBlock_51067161149791
// MI455X (gfx1250) — matrix operand order repaired and checked by reading, not compiled, not run on hardware
//
#include <hip/hip_runtime.h>
#include <math.h>

// ---------------------------------------------------------------------------
// Model constants
// ---------------------------------------------------------------------------
#define S_   512
#define D_   2048
#define H_   16
#define HD_  128
#define E_   64
#define I_   512
#define G_   8
#define TKG_ 4
#define TK_  6
#define SHI_ 1024
#define ROUTE_SCALE_ 2.5f
#define EPS_ 1e-5f

typedef __bf16 bf16;
typedef __attribute__((ext_vector_type(16))) __bf16 v16bf;
typedef __attribute__((ext_vector_type(8)))  __bf16 v8bf;
typedef __attribute__((ext_vector_type(8)))  float  v8f;

#define LSA 40   // LDS row stride in bf16 elements (80B: 16B-aligned, bank-skewed)

// ---------------------------------------------------------------------------
// WMMA fragment loaders (layouts per CDNA5 ISA 7.12.2, wave32)
// ---------------------------------------------------------------------------
// A fragment, 16x32 bf16 tile in LDS (row-major, stride LSA):
//   lanes 0-15: row M=lane, K in {0..7} (v0-3) and {16..23} (v4-7)
//   lanes16-31: row M=lane-16, K in {8..15} and {24..31}
__device__ __forceinline__ v16bf load_a_frag(const bf16* base, int lane) {
    int r  = lane & 15;
    int kh = (lane >> 4) << 3;                 // 0 or 8
    const v8bf* p0 = (const v8bf*)(base + r * LSA + kh);
    const v8bf* p1 = (const v8bf*)(base + r * LSA + kh + 16);
    v8bf lo = *p0, hi = *p1;
    v16bf f;
#pragma unroll
    for (int i = 0; i < 8; ++i) { f[i] = lo[i]; f[i + 8] = hi[i]; }
    return f;
}

// B fragment, 32x16 (KxN). LDS tile stored as [n][k] row-major, stride LSA:
//   lanes 0-15: col N=lane, K in {0..7} and {16..23}; lanes 16-31: col N=lane-16, K in {8..15} and {24..31}
__device__ __forceinline__ v16bf load_b_frag(const bf16* base, int lane) {
    int r  = lane & 15;
    int kb = (lane >> 4) << 3;                 // 0 or 8
    const v8bf* p0 = (const v8bf*)(base + r * LSA + kb);
    const v8bf* p1 = (const v8bf*)(base + r * LSA + kb + 16);
    v8bf lo = *p0, hi = *p1;
    v16bf f;
#pragma unroll
    for (int i = 0; i < 8; ++i) { f[i] = lo[i]; f[i + 8] = hi[i]; }
    return f;
}

// ---------------------------------------------------------------------------
// Generic GEMM:  C[M,N] = A[M,K](bf16) * B^T  (B given as [N,K] unless BTRANS,
// then [K,N] bf16 with N a multiple of 16).  Block tile 64x128x32, 256 thr =
// 8 waves, each wave 16x64 via 4 v_wmma_f32_16x16x32_bf16 accumulators.
// Register + LDS ping-pong double buffering: next tile's global loads are in
// flight while the current tile's WMMAs execute; one barrier per K-step.
// EPI: 0 = f32 store, 1 = bf16 store, 2 = f32 store + residual, 3 = sigmoid f32
// ---------------------------------------------------------------------------
template <typename BT, bool BTRANS, int EPI>
__global__ __launch_bounds__(256) void gemm_k(
    const bf16* __restrict__ A, int lda,
    const BT* __restrict__ B, int ldb,
    void* __restrict__ Cv, int ldc,
    const float* __restrict__ res,
    int M, int N, int K)
{
    __shared__ bf16 As[2][64 * LSA];
    __shared__ bf16 Bs[2][128 * LSA];

    const int m0   = blockIdx.y * 64;
    const int n0   = blockIdx.x * 128;
    const int tid  = threadIdx.x;
    const int lane = tid & 31;
    const int wave = tid >> 5;
    const int wm   = (wave & 3) * 16;
    const int wn   = (wave >> 2) * 64;

    const int ar = tid >> 2, ac = (tid & 3) * 8;   // A staging: 64x32
    const int br = tid >> 1, bc = (tid & 1) * 16;  // B staging (N-major)
    const int tk = tid >> 3, tn = (tid & 7) * 16;  // B staging (transposed)

    v8f acc[4] = {};

    v8bf  aR;
    float bRf[16];
    v8bf  bRb0, bRb1;

    auto fetch = [&](int k0) {
        aR = *(const v8bf*)(A + (size_t)(m0 + ar) * lda + k0 + ac);
        if constexpr (!BTRANS) {
            if (n0 + br < N) {
                const BT* src = B + (size_t)(n0 + br) * ldb + k0 + bc;
#pragma unroll
                for (int i = 0; i < 16; ++i) bRf[i] = (float)src[i];
            } else {
#pragma unroll
                for (int i = 0; i < 16; ++i) bRf[i] = 0.0f;
            }
        } else {
            const bf16* src = (const bf16*)B + (size_t)(k0 + tk) * ldb + n0 + tn;
            bRb0 = *(const v8bf*)src;
            bRb1 = *(const v8bf*)(src + 8);
        }
    };

    auto commit = [&](int buf) {
        *(v8bf*)&As[buf][ar * LSA + ac] = aR;
        if constexpr (!BTRANS) {
#pragma unroll
            for (int i = 0; i < 16; ++i) Bs[buf][br * LSA + bc + i] = (bf16)bRf[i];
        } else {
#pragma unroll
            for (int i = 0; i < 8; ++i) {
                Bs[buf][(tn + i) * LSA + tk]     = bRb0[i];
                Bs[buf][(tn + 8 + i) * LSA + tk] = bRb1[i];
            }
        }
    };

    auto compute = [&](int buf) {
        v16bf a = load_a_frag(&As[buf][wm * LSA], lane);
#pragma unroll
        for (int t = 0; t < 4; ++t) {
            v16bf b = load_b_frag(&Bs[buf][(wn + t * 16) * LSA], lane);
            acc[t] = __builtin_amdgcn_wmma_f32_16x16x32_bf16(
                false, a, false, b, (short)0, acc[t], false, false);
        }
    };

    fetch(0);
    commit(0);
    __syncthreads();

    const int nk = K / 32;
    for (int i = 0; i < nk; ++i) {
        const int kn = (i + 1) * 32;
        if (kn < K) {
            fetch(kn);                                   // loads in flight...
            if constexpr (!BTRANS) {                     // prime L2 one tile ahead
                if (kn + 32 < K && n0 + br < N)
                    __builtin_prefetch(B + (size_t)(n0 + br) * ldb + kn + 32, 0, 1);
            }
        }
        compute(i & 1);                                  // ...while WMMAs run
        if (kn < K) commit((i + 1) & 1);                 // wait lands after WMMAs
        __syncthreads();
    }

    // C layout: VGPR j -> M = j (lanes 0-15) / j+8 (lanes 16-31); N = lane&15
    int cn = lane & 15, mh = (lane >> 4) * 8;
#pragma unroll
    for (int t = 0; t < 4; ++t) {
        int ncol = n0 + wn + t * 16 + cn;
        if (ncol >= N) continue;
#pragma unroll
        for (int j = 0; j < 8; ++j) {
            int mrow = m0 + wm + mh + j;
            if (mrow >= M) continue;
            size_t ci = (size_t)mrow * ldc + ncol;
            float v = acc[t][j];
            if constexpr (EPI == 0) ((float*)Cv)[ci] = v;
            else if constexpr (EPI == 1) ((bf16*)Cv)[ci] = (bf16)v;
            else if constexpr (EPI == 2) ((float*)Cv)[ci] = v + res[ci];
            else ((float*)Cv)[ci] = 1.0f / (1.0f + __expf(-v));
        }
    }
}

// ---------------------------------------------------------------------------
// MoE expert up-projection: gathered A rows, fused W1/W3, a = silu(h1)*h3,
// same double-buffered pipeline.
// ---------------------------------------------------------------------------
__global__ __launch_bounds__(256) void expert_up_k(
    const bf16* __restrict__ t_bf,               // [512, 2048]
    const float* __restrict__ w1,                // [E, I, D]
    const float* __restrict__ w3,                // [E, I, D]
    const int* __restrict__ counts,
    const int* __restrict__ tok,                 // [E, 512]
    bf16* __restrict__ a_out)                    // [E, 512, I]
{
    const int e   = blockIdx.z;
    const int cnt = counts[e];
    const int m0  = blockIdx.y * 64;
    if (m0 >= cnt) return;
    const int n0  = blockIdx.x * 128;

    __shared__ bf16 As[2][64 * LSA];
    __shared__ bf16 B1[2][128 * LSA];
    __shared__ bf16 B3[2][128 * LSA];

    const int tid = threadIdx.x, lane = tid & 31, wave = tid >> 5;
    const int wm = (wave & 3) * 16, wn = (wave >> 2) * 64;
    const int ar = tid >> 2, ac = (tid & 3) * 8;
    const int br = tid >> 1, bc = (tid & 1) * 16;
    const float* w1e = w1 + (size_t)e * I_ * D_;
    const float* w3e = w3 + (size_t)e * I_ * D_;
    const int* tl = tok + e * S_;
    const int am = m0 + ar;
    const size_t arow = (am < cnt) ? (size_t)tl[am] * D_ : 0;
    const bool avalid = (am < cnt);

    v8f acc1[4] = {}, acc3[4] = {};
    v8bf aR;
    float b1R[16], b3R[16];

    auto fetch = [&](int k0) {
        if (avalid) aR = *(const v8bf*)&t_bf[arow + k0 + ac];
        else {
#pragma unroll
            for (int i = 0; i < 8; ++i) aR[i] = (bf16)0.0f;
        }
        const float* s1 = w1e + (size_t)(n0 + br) * D_ + k0 + bc;
        const float* s3 = w3e + (size_t)(n0 + br) * D_ + k0 + bc;
#pragma unroll
        for (int i = 0; i < 16; ++i) { b1R[i] = s1[i]; b3R[i] = s3[i]; }
    };
    auto commit = [&](int buf) {
        *(v8bf*)&As[buf][ar * LSA + ac] = aR;
#pragma unroll
        for (int i = 0; i < 16; ++i) {
            B1[buf][br * LSA + bc + i] = (bf16)b1R[i];
            B3[buf][br * LSA + bc + i] = (bf16)b3R[i];
        }
    };
    auto compute = [&](int buf) {
        v16bf a = load_a_frag(&As[buf][wm * LSA], lane);
#pragma unroll
        for (int t = 0; t < 4; ++t) {
            v16bf b1 = load_b_frag(&B1[buf][(wn + t * 16) * LSA], lane);
            acc1[t] = __builtin_amdgcn_wmma_f32_16x16x32_bf16(
                false, a, false, b1, (short)0, acc1[t], false, false);
            v16bf b3 = load_b_frag(&B3[buf][(wn + t * 16) * LSA], lane);
            acc3[t] = __builtin_amdgcn_wmma_f32_16x16x32_bf16(
                false, a, false, b3, (short)0, acc3[t], false, false);
        }
    };

    fetch(0); commit(0); __syncthreads();
    for (int i = 0; i < D_ / 32; ++i) {
        const int kn = (i + 1) * 32;
        if (kn < D_) {
            fetch(kn);
            if (kn + 32 < D_) {
                __builtin_prefetch(w1e + (size_t)(n0 + br) * D_ + kn + 32, 0, 1);
                __builtin_prefetch(w3e + (size_t)(n0 + br) * D_ + kn + 32, 0, 1);
            }
        }
        compute(i & 1);
        if (kn < D_) commit((i + 1) & 1);
        __syncthreads();
    }

    int cn = lane & 15, mh = (lane >> 4) * 8;
#pragma unroll
    for (int t = 0; t < 4; ++t) {
        int ncol = n0 + wn + t * 16 + cn;
#pragma unroll
        for (int j = 0; j < 8; ++j) {
            int mrow = m0 + wm + mh + j;
            if (mrow >= cnt) continue;
            float h1 = acc1[t][j], h3 = acc3[t][j];
            float sl = h1 / (1.0f + __expf(-h1));
            a_out[(size_t)e * S_ * I_ + (size_t)mrow * I_ + ncol] = (bf16)(sl * h3);
        }
    }
}

// ---------------------------------------------------------------------------
// MoE expert down-projection: y = a * W2^T, scaled by gate weight,
// atomically accumulated into accum[token, :]
// ---------------------------------------------------------------------------
__global__ __launch_bounds__(256) void expert_down_k(
    const bf16* __restrict__ a_in,               // [E, 512, I]
    const float* __restrict__ w2,                // [E, D, I]
    const int* __restrict__ counts,
    const int* __restrict__ tok,
    const float* __restrict__ wtl,
    float* __restrict__ accum)                   // [512, 2048]
{
    const int e   = blockIdx.z;
    const int cnt = counts[e];
    const int m0  = blockIdx.y * 64;
    if (m0 >= cnt) return;
    const int n0  = blockIdx.x * 128;

    __shared__ bf16 As[2][64 * LSA];
    __shared__ bf16 Bs[2][128 * LSA];

    const int tid = threadIdx.x, lane = tid & 31, wave = tid >> 5;
    const int wm = (wave & 3) * 16, wn = (wave >> 2) * 64;
    const int ar = tid >> 2, ac = (tid & 3) * 8;
    const int br = tid >> 1, bc = (tid & 1) * 16;
    const bf16* Ae = a_in + (size_t)e * S_ * I_;
    const float* Be = w2 + (size_t)e * D_ * I_;
    const bool avalid = (m0 + ar < cnt);

    v8f acc[4] = {};
    v8bf aR;
    float bR[16];

    auto fetch = [&](int k0) {
        if (avalid) aR = *(const v8bf*)&Ae[(size_t)(m0 + ar) * I_ + k0 + ac];
        else {
#pragma unroll
            for (int i = 0; i < 8; ++i) aR[i] = (bf16)0.0f;
        }
        const float* src = Be + (size_t)(n0 + br) * I_ + k0 + bc;
#pragma unroll
        for (int i = 0; i < 16; ++i) bR[i] = src[i];
    };
    auto commit = [&](int buf) {
        *(v8bf*)&As[buf][ar * LSA + ac] = aR;
#pragma unroll
        for (int i = 0; i < 16; ++i) Bs[buf][br * LSA + bc + i] = (bf16)bR[i];
    };
    auto compute = [&](int buf) {
        v16bf a = load_a_frag(&As[buf][wm * LSA], lane);
#pragma unroll
        for (int t = 0; t < 4; ++t) {
            v16bf b = load_b_frag(&Bs[buf][(wn + t * 16) * LSA], lane);
            acc[t] = __builtin_amdgcn_wmma_f32_16x16x32_bf16(
                false, a, false, b, (short)0, acc[t], false, false);
        }
    };

    fetch(0); commit(0); __syncthreads();
    for (int i = 0; i < I_ / 32; ++i) {
        const int kn = (i + 1) * 32;
        if (kn < I_) {
            fetch(kn);
            if (kn + 32 < I_)
                __builtin_prefetch(Be + (size_t)(n0 + br) * I_ + kn + 32, 0, 1);
        }
        compute(i & 1);
        if (kn < I_) commit((i + 1) & 1);
        __syncthreads();
    }

    int cn = lane & 15, mh = (lane >> 4) * 8;
#pragma unroll
    for (int t = 0; t < 4; ++t) {
        int ncol = n0 + wn + t * 16 + cn;
#pragma unroll
        for (int j = 0; j < 8; ++j) {
            int mrow = m0 + wm + mh + j;
            if (mrow >= cnt) continue;
            int   tkn = tok[e * S_ + mrow];
            float sc  = wtl[e * S_ + mrow];
            atomicAdd(&accum[(size_t)tkn * D_ + ncol], acc[t][j] * sc);
        }
    }
}

// ---------------------------------------------------------------------------
// Elementwise / reduction kernels
// ---------------------------------------------------------------------------
__global__ __launch_bounds__(256) void rmsnorm_k(
    const float* __restrict__ x, const float* __restrict__ w,
    bf16* __restrict__ out)
{
    const int row = blockIdx.x, tid = threadIdx.x;
    const float* xr = x + (size_t)row * D_;
    __shared__ float red[256];
    float ss = 0.0f;
    for (int i = tid; i < D_; i += 256) { float v = xr[i]; ss += v * v; }
    red[tid] = ss; __syncthreads();
    for (int o = 128; o > 0; o >>= 1) {
        if (tid < o) red[tid] += red[tid + o];
        __syncthreads();
    }
    float rs = rsqrtf(red[0] / (float)D_ + EPS_);
    for (int i = tid; i < D_; i += 256)
        out[(size_t)row * D_ + i] = (bf16)(xr[i] * rs * w[i]);
}

__global__ __launch_bounds__(256) void rope_k(
    const float* __restrict__ q, const float* __restrict__ k,
    const float* __restrict__ fc, const float* __restrict__ fs,
    bf16* __restrict__ qo, bf16* __restrict__ ko)
{
    int gid = blockIdx.x * 256 + threadIdx.x;       // S*H*(HD/2)
    if (gid >= S_ * H_ * (HD_ / 2)) return;
    int p = gid & 63, hh = (gid >> 6) & 15, s = gid >> 10;
    size_t base = (size_t)s * D_ + hh * HD_ + p * 2;
    float c = fc[s * 64 + p], sn = fs[s * 64 + p];
    float qr = q[base], qi = q[base + 1];
    float kr = k[base], ki = k[base + 1];
    const float scale = 0.088388347648318447f;      // 1/sqrt(HD)
    qo[base]     = (bf16)((qr * c - qi * sn) * scale);
    qo[base + 1] = (bf16)((qr * sn + qi * c) * scale);
    ko[base]     = (bf16)(kr * c - ki * sn);
    ko[base + 1] = (bf16)(kr * sn + ki * c);
}

__global__ __launch_bounds__(256) void softmax_k(
    const float* __restrict__ scores, const float* __restrict__ mask,
    bf16* __restrict__ p)
{
    const int h = blockIdx.y, row = blockIdx.x, tid = threadIdx.x;
    const float* sr = scores + ((size_t)h * S_ + row) * S_;
    const float* mr = mask + (size_t)row * S_;
    __shared__ float red[256];
    float v0 = sr[tid] + mr[tid];
    float v1 = sr[tid + 256] + mr[tid + 256];
    red[tid] = fmaxf(v0, v1); __syncthreads();
    for (int o = 128; o > 0; o >>= 1) {
        if (tid < o) red[tid] = fmaxf(red[tid], red[tid + o]);
        __syncthreads();
    }
    float M = red[0]; __syncthreads();
    float e0 = __expf(v0 - M), e1 = __expf(v1 - M);
    red[tid] = e0 + e1; __syncthreads();
    for (int o = 128; o > 0; o >>= 1) {
        if (tid < o) red[tid] += red[tid + o];
        __syncthreads();
    }
    float inv = 1.0f / red[0];
    bf16* pr = p + ((size_t)h * S_ + row) * S_;
    pr[tid]       = (bf16)(e0 * inv);
    pr[tid + 256] = (bf16)(e1 * inv);
}

__global__ __launch_bounds__(64) void route_k(
    const float* __restrict__ s,                  // [512, 64] sigmoid gates
    int* __restrict__ idx, float* __restrict__ wts)
{
    const int t = blockIdx.x, e = threadIdx.x;
    __shared__ float sv[E_], gs[G_], mv[E_];
    __shared__ int keep[G_], six[TK_];
    __shared__ float sw[TK_];
    sv[e] = s[(size_t)t * E_ + e];
    if (e < G_) keep[e] = 0;
    __syncthreads();
    if (e < G_) {
        float m = -INFINITY;
        for (int j = 0; j < 8; ++j) m = fmaxf(m, sv[e * 8 + j]);
        gs[e] = m;
    }
    __syncthreads();
    if (e == 0) {                                 // top-4 groups (first-on-tie)
        float tmp[G_];
        for (int g = 0; g < G_; ++g) tmp[g] = gs[g];
        for (int r = 0; r < TKG_; ++r) {
            int best = 0; float bv = -INFINITY;
            for (int g = 0; g < G_; ++g)
                if (tmp[g] > bv) { bv = tmp[g]; best = g; }
            keep[best] = 1; tmp[best] = -INFINITY;
        }
    }
    __syncthreads();
    mv[e] = keep[e >> 3] ? sv[e] : -INFINITY;
    __syncthreads();
    if (e == 0) {                                 // top-6 experts + weights
        float sum = 0.0f;
        for (int r = 0; r < TK_; ++r) {
            int best = 0; float bv = -INFINITY;
            for (int j = 0; j < E_; ++j)
                if (mv[j] > bv) { bv = mv[j]; best = j; }
            six[r] = best; sw[r] = sv[best]; sum += sv[best];
            mv[best] = -INFINITY;
        }
        float inv = ROUTE_SCALE_ / sum;
        for (int r = 0; r < TK_; ++r) sw[r] *= inv;
    }
    __syncthreads();
    if (e < TK_) { idx[t * TK_ + e] = six[e]; wts[t * TK_ + e] = sw[e]; }
}

__global__ void zero_counts_k(int* c) { if (threadIdx.x < E_) c[threadIdx.x] = 0; }

__global__ __launch_bounds__(32) void scatter_k(
    const int* __restrict__ idx, const float* __restrict__ wts,
    int* __restrict__ counts, int* __restrict__ tok, float* __restrict__ wtl)
{
    const int t = blockIdx.x, s = threadIdx.x;
    if (s < TK_) {
        int e = idx[t * TK_ + s];
        int pos = atomicAdd(&counts[e], 1);
        tok[e * S_ + pos] = t;
        wtl[e * S_ + pos] = wts[t * TK_ + s];
    }
}

__global__ __launch_bounds__(256) void silu_mul_k(
    const float* __restrict__ g1, const float* __restrict__ g3,
    bf16* __restrict__ u, int n)
{
    int i = blockIdx.x * 256 + threadIdx.x;
    if (i < n) {
        float a = g1[i];
        u[i] = (bf16)((a / (1.0f + __expf(-a))) * g3[i]);
    }
}

__global__ __launch_bounds__(256) void final_add_k(
    const float* __restrict__ x1, const float* __restrict__ acc,
    float* __restrict__ out, int n)
{
    int i = blockIdx.x * 256 + threadIdx.x;
    if (i < n) out[i] = x1[i] + acc[i];
}

// ---------------------------------------------------------------------------
// Host-side orchestration
// ---------------------------------------------------------------------------
extern "C" void kernel_launch(void* const* d_in, const int* in_sizes, int n_in,
                              void* d_out, int out_size, void* d_ws, size_t ws_size,
                              hipStream_t stream)
{
    const float* x       = (const float*)d_in[0];
    const float* fcos    = (const float*)d_in[1];
    const float* fsin    = (const float*)d_in[2];
    const float* mask    = (const float*)d_in[3];
    const float* anw     = (const float*)d_in[4];
    const float* fnw     = (const float*)d_in[5];
    const float* wq      = (const float*)d_in[6];
    const float* wk      = (const float*)d_in[7];
    const float* wv      = (const float*)d_in[8];
    const float* wo      = (const float*)d_in[9];
    const float* gate_w  = (const float*)d_in[10];
    const float* exp_w1  = (const float*)d_in[11];
    const float* exp_w2  = (const float*)d_in[12];
    const float* exp_w3  = (const float*)d_in[13];
    const float* sh_w1   = (const float*)d_in[14];
    const float* sh_w2   = (const float*)d_in[15];
    const float* sh_w3   = (const float*)d_in[16];
    float* out = (float*)d_out;

    char* p = (char*)d_ws;
    auto alloc = [&](size_t bytes) -> void* {
        void* r = (void*)p;
        p += (bytes + 255) & ~(size_t)255;
        return r;
    };
    bf16*  h_bf    = (bf16*)alloc((size_t)S_ * D_ * 2);
    float* q_f     = (float*)alloc((size_t)S_ * D_ * 4);
    float* k_f     = (float*)alloc((size_t)S_ * D_ * 4);
    bf16*  q_bf    = (bf16*)alloc((size_t)S_ * D_ * 2);
    bf16*  k_bf    = (bf16*)alloc((size_t)S_ * D_ * 2);
    bf16*  v_bf    = (bf16*)alloc((size_t)S_ * D_ * 2);
    float* scores  = (float*)alloc((size_t)H_ * S_ * S_ * 4);
    bf16*  p_bf    = (bf16*)alloc((size_t)H_ * S_ * S_ * 2);
    bf16*  attn_bf = (bf16*)alloc((size_t)S_ * D_ * 2);
    float* x1      = (float*)alloc((size_t)S_ * D_ * 4);
    bf16*  t_bf    = (bf16*)alloc((size_t)S_ * D_ * 2);
    float* sgate   = (float*)alloc((size_t)S_ * E_ * 4);
    int*   idx     = (int*)alloc((size_t)S_ * TK_ * 4);
    float* wts     = (float*)alloc((size_t)S_ * TK_ * 4);
    int*   counts  = (int*)alloc((size_t)E_ * 4);
    int*   tok     = (int*)alloc((size_t)E_ * S_ * 4);
    float* wtl     = (float*)alloc((size_t)E_ * S_ * 4);
    float* g1      = (float*)alloc((size_t)S_ * SHI_ * 4);
    float* g3      = (float*)alloc((size_t)S_ * SHI_ * 4);
    bf16*  u_bf    = (bf16*)alloc((size_t)S_ * SHI_ * 2);
    bf16*  a_bf    = (bf16*)alloc((size_t)E_ * S_ * I_ * 2);
    float* accum   = (float*)alloc((size_t)S_ * D_ * 4);

    const dim3 blk(256);

    // 1) attn rmsnorm -> h (bf16)
    rmsnorm_k<<<dim3(S_), blk, 0, stream>>>(x, anw, h_bf);

    // 2) Q, K (f32 for RoPE), V (bf16 directly)
    gemm_k<float, false, 0><<<dim3(D_/128, S_/64), blk, 0, stream>>>(
        h_bf, D_, wq, D_, q_f, D_, nullptr, S_, D_, D_);
    gemm_k<float, false, 0><<<dim3(D_/128, S_/64), blk, 0, stream>>>(
        h_bf, D_, wk, D_, k_f, D_, nullptr, S_, D_, D_);
    gemm_k<float, false, 1><<<dim3(D_/128, S_/64), blk, 0, stream>>>(
        h_bf, D_, wv, D_, v_bf, D_, nullptr, S_, D_, D_);

    // 3) RoPE (q pre-scaled by 1/sqrt(HD)) -> bf16
    rope_k<<<dim3((S_*H_*(HD_/2) + 255)/256), blk, 0, stream>>>(
        q_f, k_f, fcos, fsin, q_bf, k_bf);

    // 4) attention scores per head: S = q . k^T
    for (int h = 0; h < H_; ++h)
        gemm_k<bf16, false, 0><<<dim3(S_/128, S_/64), blk, 0, stream>>>(
            q_bf + h*HD_, D_, k_bf + h*HD_, D_,
            scores + (size_t)h*S_*S_, S_, nullptr, S_, S_, HD_);

    // 5) masked softmax -> P (bf16)
    softmax_k<<<dim3(S_, H_), blk, 0, stream>>>(scores, mask, p_bf);

    // 6) attn = P * V per head (B transposed: V is [K=S, N=HD])
    for (int h = 0; h < H_; ++h)
        gemm_k<bf16, true, 1><<<dim3(1, S_/64), blk, 0, stream>>>(
            p_bf + (size_t)h*S_*S_, S_, v_bf + h*HD_, D_,
            attn_bf + h*HD_, D_, nullptr, S_, HD_, S_);

    // 7) x1 = x + attn * wo^T
    gemm_k<float, false, 2><<<dim3(D_/128, S_/64), blk, 0, stream>>>(
        attn_bf, D_, wo, D_, x1, D_, x, S_, D_, D_);

    // 8) ffn rmsnorm -> t (bf16)
    rmsnorm_k<<<dim3(S_), blk, 0, stream>>>(x1, fnw, t_bf);

    // 9) gate logits -> sigmoid
    gemm_k<float, false, 3><<<dim3(1, S_/64), blk, 0, stream>>>(
        t_bf, D_, gate_w, D_, sgate, E_, nullptr, S_, E_, D_);

    // 10) routing: group top-4, expert top-6, normalized weights
    route_k<<<dim3(S_), dim3(64), 0, stream>>>(sgate, idx, wts);
    zero_counts_k<<<dim3(1), dim3(64), 0, stream>>>(counts);
    scatter_k<<<dim3(S_), dim3(32), 0, stream>>>(idx, wts, counts, tok, wtl);

    // 11) shared FFN: accum = (silu(t*w1^T) * (t*w3^T)) * w2^T
    gemm_k<float, false, 0><<<dim3(SHI_/128, S_/64), blk, 0, stream>>>(
        t_bf, D_, sh_w1, D_, g1, SHI_, nullptr, S_, SHI_, D_);
    gemm_k<float, false, 0><<<dim3(SHI_/128, S_/64), blk, 0, stream>>>(
        t_bf, D_, sh_w3, D_, g3, SHI_, nullptr, S_, SHI_, D_);
    silu_mul_k<<<dim3((S_*SHI_ + 255)/256), blk, 0, stream>>>(
        g1, g3, u_bf, S_*SHI_);
    gemm_k<float, false, 0><<<dim3(D_/128, S_/64), blk, 0, stream>>>(
        u_bf, SHI_, sh_w2, SHI_, accum, D_, nullptr, S_, D_, SHI_);

    // 12) routed experts: sparse grouped GEMMs, atomic add onto accum
    expert_up_k<<<dim3(I_/128, S_/64, E_), blk, 0, stream>>>(
        t_bf, exp_w1, exp_w3, counts, tok, a_bf);
    expert_down_k<<<dim3(D_/128, S_/64, E_), blk, 0, stream>>>(
        a_bf, exp_w2, counts, tok, wtl, accum);

    // 13) out = x1 + (routed + shared)
    final_add_k<<<dim3((S_*D_ + 255)/256), blk, 0, stream>>>(
        x1, accum, out, S_*D_);
}
